// TemporalRefiner_3178275799110
// MI455X (gfx1250) — matrix operand order repaired and checked by reading, not compiled, not run on hardware
//
#include <hip/hip_runtime.h>

// Bidirectional LSTM (H=32, I=4, B=256, T=4096) + linear head, fused.
// Strategy: gates computed transposed (128 gates x 16 batch) per wave:
//   G_tile = WMMA_f32_16x16x4_f32 (A=Wih tile, B=x^T, C=bias)       // K=4 == I
//   G_tile = WMMA_f32_16x16x32_f16(A=Whh tile, B=h^T f16, C=G_tile) // K=32 == H
// h D-layout -> next-step B-layout via ds_swizzle SWAPX16 (no LDS memory).
// x load software-pipelined by one iteration to keep it off the recurrence
// critical path. 32 single-wave blocks (16 batch tiles x 2 directions).

namespace {
constexpr int T_LEN = 4096;
constexpr int B_SZ  = 256;
constexpr int H_SZ  = 32;
}

typedef __attribute__((ext_vector_type(16))) _Float16 v16h;
typedef __attribute__((ext_vector_type(8)))  float    v8f;
typedef __attribute__((ext_vector_type(2)))  float    v2f;
typedef __attribute__((ext_vector_type(8)))  unsigned v8u;

#if __has_builtin(__builtin_amdgcn_tanhf)
// CDNA5 V_TANH_F32: 1 TRANS op per activation
__device__ __forceinline__ float tanh_f(float x) {
  return __builtin_amdgcn_tanhf(x);
}
__device__ __forceinline__ float sigm(float x) {
  return __builtin_fmaf(0.5f, __builtin_amdgcn_tanhf(0.5f * x), 0.5f);
}
#else
__device__ __forceinline__ float sigm(float x) {
  return __builtin_amdgcn_rcpf(1.0f + __expf(-x));
}
__device__ __forceinline__ float tanh_f(float x) {
  return 2.0f * __builtin_amdgcn_rcpf(1.0f + __expf(-2.0f * x)) - 1.0f;
}
#endif

// lane l <-> lane l^16 (SWAPX16): group-of-32 swizzle, and=0x1f, xor=0x10
__device__ __forceinline__ unsigned swap16(unsigned x) {
  return (unsigned)__builtin_amdgcn_ds_swizzle((int)x, 0x401f);
}

// one 16-gate x 16-batch tile: xproj (K=4 f32 WMMA, bias in C) then +Whh*h (K=32 f16 WMMA)
__device__ __forceinline__ v8f gate_tile(v2f aw, v2f bx, v8f cbias, v16h ah, v16h bh) {
  v8f c = __builtin_amdgcn_wmma_f32_16x16x4_f32(false, aw, false, bx, (short)0, cbias,
                                                false, false);
  return __builtin_amdgcn_wmma_f32_16x16x32_f16(false, ah, false, bh, (short)0, c,
                                                false, false);
}

__global__ void __launch_bounds__(32)
bilstm_fused_kernel(const float* __restrict__ x,
                    const float* __restrict__ WihF, const float* __restrict__ WhhF,
                    const float* __restrict__ bF,
                    const float* __restrict__ WihB, const float* __restrict__ WhhB,
                    const float* __restrict__ bB,
                    const float* __restrict__ Wh,
                    float* __restrict__ pF, float* __restrict__ pB)
{
  const int  lane  = (int)(threadIdx.x & 31u);
  const bool hi    = lane >= 16;
  const int  m     = lane & 15;
  const int  dir   = (int)(blockIdx.x >> 4);      // 0 = forward, 1 = backward
  const int  bbase = (int)(blockIdx.x & 15u) << 4;

  const float* Wih  = dir ? WihB : WihF;
  const float* Whh  = dir ? WhhB : WhhF;
  const float* bias = dir ? bB   : bF;
  const float* wv   = Wh + dir * H_SZ;            // head weights for this direction
  float*       pout = dir ? pB : pF;

  // per-lane fixed base into x: batch row (bbase+m), fp offset (hi?2:0)
  const float* xlane = x + ((size_t)(bbase + m) * T_LEN << 2) + (hi ? 2 : 0);

  // ---- preload constant operands into registers ----
  // A(Wih) tile, f32 16x4 A layout: VGPR0 = K0 (lanes<16) / K2 (lanes>=16); VGPR1 = K1/K3
  v2f  aWih[8];
  // A(Whh) tile, f16 16x32 A in this code's contiguous k order (same as B): elems 0..15 = K k0..k0+15 (k0 = hi?16:0)
  v16h aWhh[8];
  // bias in C/D layout: [r] = bias[tile*16 + r + 8*hi]
  v8f  cb[8];
  {
    const int k0 = hi ? 16 : 0;
#pragma unroll
    for (int j = 0; j < 8; ++j) {
      const int g = j * 16 + m;
      aWih[j].x = Wih[g * 4 + (hi ? 2 : 0)];
      aWih[j].y = Wih[g * 4 + (hi ? 3 : 1)];
#pragma unroll
      for (int e = 0; e < 8; ++e) {
        aWhh[j][e]     = (_Float16)Whh[g * 32 + k0 + e];
        aWhh[j][8 + e] = (_Float16)Whh[g * 32 + k0 + 8 + e];
      }
#pragma unroll
      for (int r = 0; r < 8; ++r)
        cb[j][r] = bias[j * 16 + r + (hi ? 8 : 0)];
    }
  }

  // head weights, matching h tile layout: hlo[r] = h[hidden r+8*hi], hhi[r] = h[16+r+8*hi]
  v8f wlo, whi_;
#pragma unroll
  for (int r = 0; r < 8; ++r) {
    wlo[r]  = wv[r + (hi ? 8 : 0)];
    whi_[r] = wv[16 + r + (hi ? 8 : 0)];
  }

  v8f clo = {};
  v8f chi = {};
  v8u bhu = {};   // packed f16 B-operand of h (starts at h0 = 0)

  // x^T B-operand, f32 4x16 B layout: V0 = K0 (lo lanes)/K2 (hi lanes), V1 = K1/K3
  // Software pipeline: bx holds data for step s; issue load for step s+1 up front.
  v2f bx;
  {
    const int t0 = dir ? (T_LEN - 1) : 0;
    bx.x = xlane[(size_t)t0 << 2];
    bx.y = xlane[((size_t)t0 << 2) + 1];
  }

  for (int s = 0; s < T_LEN; ++s) {
    const int t = dir ? (T_LEN - 1 - s) : s;
    // next-step x load issued first; clamped index keeps the load unconditional
    int tn = dir ? (T_LEN - 2 - s) : (s + 1);
    tn = tn < 0 ? 0 : (tn >= T_LEN ? T_LEN - 1 : tn);
    v2f bx_next;
    bx_next.x = xlane[(size_t)tn << 2];
    bx_next.y = xlane[((size_t)tn << 2) + 1];

    const v16h Bh = __builtin_bit_cast(v16h, bhu);

    // gate order i,f,g,o; tiles 2k,2k+1 = hidden 0-15 / 16-31 of gate k
    // --- hidden 0..15 half ---
    v8f Gi = gate_tile(aWih[0], bx, cb[0], aWhh[0], Bh);
    v8f Gf = gate_tile(aWih[2], bx, cb[2], aWhh[2], Bh);
    v8f Gg = gate_tile(aWih[4], bx, cb[4], aWhh[4], Bh);
    v8f Go = gate_tile(aWih[6], bx, cb[6], aWhh[6], Bh);
    v8f hlo;
#pragma unroll
    for (int r = 0; r < 8; ++r) {
      float iv = sigm(Gi[r]);
      float fv = sigm(Gf[r]);
      float gv = tanh_f(Gg[r]);
      float ov = sigm(Go[r]);
      float c  = fv * clo[r] + iv * gv;
      clo[r] = c;
      hlo[r] = ov * tanh_f(c);
    }
    // --- hidden 16..31 half ---
    Gi = gate_tile(aWih[1], bx, cb[1], aWhh[1], Bh);
    Gf = gate_tile(aWih[3], bx, cb[3], aWhh[3], Bh);
    Gg = gate_tile(aWih[5], bx, cb[5], aWhh[5], Bh);
    Go = gate_tile(aWih[7], bx, cb[7], aWhh[7], Bh);
    v8f hhi;
#pragma unroll
    for (int r = 0; r < 8; ++r) {
      float iv = sigm(Gi[r]);
      float fv = sigm(Gf[r]);
      float gv = tanh_f(Gg[r]);
      float ov = sigm(Go[r]);
      float c  = fv * chi[r] + iv * gv;
      chi[r] = c;
      hhi[r] = ov * tanh_f(c);
    }

    // fused head partial: dot(h, Wh_dir) per batch column; halves combined via SWAPX16
    float p = 0.0f;
#pragma unroll
    for (int r = 0; r < 8; ++r) p += wlo[r] * hlo[r] + whi_[r] * hhi[r];
    p += __int_as_float((int)swap16((unsigned)__float_as_int(p)));
    if (!hi) pout[(size_t)(bbase + lane) * T_LEN + t] = p;

    // h (f32 D layout, halves split by +-8) -> f16 B operand (halves split by +-16):
    // pack pairs, then one SWAPX16 per pair moves the cross-half data.
    unsigned PL[4], PH[4];
#pragma unroll
    for (int j = 0; j < 4; ++j) {
      PL[j] = __builtin_bit_cast(unsigned,
                __builtin_amdgcn_cvt_pkrtz(hlo[2 * j], hlo[2 * j + 1]));
      PH[j] = __builtin_bit_cast(unsigned,
                __builtin_amdgcn_cvt_pkrtz(hhi[2 * j], hhi[2 * j + 1]));
    }
#pragma unroll
    for (int j = 0; j < 4; ++j) {
      unsigned S = hi ? PL[j] : PH[j];
      unsigned U = swap16(S);          // lo lanes get PL from l+16, hi lanes get PH from l-16
      bhu[j]     = hi ? U : PL[j];     // K = 2j,2j+1   (+16 on hi lanes)
      bhu[4 + j] = hi ? PH[j] : U;     // K = 8+2j,9+2j (+16 on hi lanes)
    }

    bx = bx_next;
  }
}

__global__ void head_combine_kernel(const float* __restrict__ pF,
                                    const float* __restrict__ pB,
                                    const float* __restrict__ bh,
                                    float* __restrict__ out, int n)
{
  int i = (int)(blockIdx.x * blockDim.x + threadIdx.x);
  if (i < n) out[i] = pF[i] + pB[i] + bh[0];
}

extern "C" void kernel_launch(void* const* d_in, const int* in_sizes, int n_in,
                              void* d_out, int out_size, void* d_ws, size_t ws_size,
                              hipStream_t stream) {
  (void)in_sizes; (void)n_in; (void)out_size; (void)ws_size;
  const float* x     = (const float*)d_in[0];
  const float* Wih_f = (const float*)d_in[1];
  const float* Whh_f = (const float*)d_in[2];
  const float* b_f   = (const float*)d_in[3];
  const float* Wih_b = (const float*)d_in[4];
  const float* Whh_b = (const float*)d_in[5];
  const float* b_b   = (const float*)d_in[6];
  const float* Wh    = (const float*)d_in[7];
  const float* bh    = (const float*)d_in[8];

  float* pF = (float*)d_ws;                        // (B,T) fwd head partials
  float* pB = pF + (size_t)B_SZ * T_LEN;           // (B,T) bwd head partials

  // 16 batch tiles x 2 directions, one wave32 per block
  bilstm_fused_kernel<<<32, 32, 0, stream>>>(x, Wih_f, Whh_f, b_f,
                                             Wih_b, Whh_b, b_b, Wh, pF, pB);

  const int n = B_SZ * T_LEN;
  head_combine_kernel<<<(n + 255) / 256, 256, 0, stream>>>(pF, pB, bh, (float*)d_out, n);
}
